// LatentODE_76776835383575
// MI455X (gfx1250) — compile-verified, matrix operand order checked by reading, not run on hardware
//
#include <hip/hip_runtime.h>
#include <hip/hip_bf16.h>

// ---------------------------------------------------------------------------
// LatentODE rollout on gfx1250 (MI455X).
//
// dims: B=1024, FA=64, FS=64, LD=128, H=512, T=64, DT=0.1
// ~211 GFLOP of chained GEMMs, ~33MB of HBM traffic -> compute bound on the
// matrix pipes.  Strategy:
//   * bf16 WMMA (v_wmma_f32_16x16x32_bf16) with f32 accumulation
//   * batch split: 64 workgroups x 16 batch rows; 4 waves/WG each owning a
//     cyclic quarter of the N dimension (N-tiles nt = ng + 4*i)
//   * activations in LDS (bf16, bank-conflict-free padded strides)
//   * weights converted once to bf16 in d_ws; streamed from the 192MB L2.
//     The weight pointer is laundered as an address_space(1) pointer inside
//     wave_gemm: blocks LICM (which caused scratch spilling) while keeping
//     the loads on the GLOBAL path (LOADcnt only) instead of FLAT
//     (LOADcnt+DScnt, 64-bit VGPR addressing).
//   * y0 / RK4 k-sum kept entirely in registers (identical (m,n)->thread
//     mapping for every N=128 GEMM), epilogues fused
// ---------------------------------------------------------------------------

#define B_DIM 1024
#define FA    64
#define FS    64
#define LD    128
#define H_DIM 512
#define T_DIM 64
#define DT_V  0.1f
#define BT    16      // batch rows per workgroup

// LDS strides (elements).  stride*2 must be 16B-aligned; stride*2 mod 256
// chosen (16 or 48 bytes) so 16 consecutive rows map to distinct bank groups.
#define SAUG  280     // aug buffer [BT][280]: cols 0..63 a_t | 64..191 z | 192..255 s
#define SY    136     // y (RK stage input) [BT][136]
#define SH    520     // hidden activations [BT][520]

// bf16 weight offsets inside d_ws (elements)
#define OFF_WAUG 0u            // 128 x 256
#define OFF_W1   32768u        // 512 x 128
#define OFF_W2   98304u        // 512 x 512
#define OFF_W3   360448u       // 128 x 512
#define OFF_WP   425984u       // 64  x 128
#define W_TOTAL  434176u

typedef __attribute__((ext_vector_type(16))) __bf16        v16bf;
typedef __attribute__((ext_vector_type(8)))  float         v8f;
typedef __attribute__((ext_vector_type(4)))  unsigned int  v4u;

// global (address_space(1)) views for weight fragment loads
typedef __attribute__((address_space(1))) const unsigned short g_ushort;
typedef __attribute__((address_space(1))) const v4u            g_v4u;

union Frag { v16bf v; v4u q[2]; };
static_assert(sizeof(Frag) == 32, "frag size");

__device__ __forceinline__ unsigned short f2bf(float f) {
  unsigned int u = __float_as_uint(f);
  u += 0x7fffu + ((u >> 16) & 1u);          // round-to-nearest-even
  return (unsigned short)(u >> 16);
}

__device__ __forceinline__ float fast_tanh(float x) {
  x = fminf(fmaxf(x, -15.f), 15.f);
  float e = __expf(2.f * x);                // v_exp_f32 path
  return __fdividef(e - 1.f, e + 1.f);
}

// One 16-row strip x NT 16-col tiles GEMM, K contraction in steps of 32.
// A: bf16 in LDS, row-major with stride lda.  W: bf16 (N,K) row-major in
// global (B[k][n] = W[n][k]); per fragment each lane reads two contiguous
// 16-byte chunks, matching the documented 16-bit WMMA operand layout.
template<int K, int NT>
__device__ __forceinline__ void wave_gemm(const unsigned short* __restrict__ A, int lda,
                                          const unsigned short* __restrict__ W,
                                          int nTile0, v8f (&acc)[NT]) {
  const int lane = threadIdx.x & 31;
  const int l15  = lane & 15;
  const int g    = lane >> 4;
  const unsigned short* arow = A + l15 * lda;

  // Global-address-space view of the weights, laundered so LICM cannot hoist
  // the (loop-invariant) fragment loads out of the time loop (which caused
  // massive scratch spilling).  AS(1) keeps the loads on global_load_b128
  // (LOADcnt only) instead of flat_load (LOADcnt+DScnt + 64-bit vaddr math).
  g_ushort* Wg = (g_ushort*)W;
  asm volatile("" : "+s"(Wg));

  const v8f vzero = {0.f,0.f,0.f,0.f,0.f,0.f,0.f,0.f};
#pragma unroll
  for (int nt = 0; nt < NT; ++nt) acc[nt] = vzero;

#pragma unroll 2
  for (int kt = 0; kt < K; kt += 32) {
    Frag a;
    a.q[0] = *(const v4u*)(arow + kt +      8 * g);
    a.q[1] = *(const v4u*)(arow + kt + 16 + 8 * g);
#pragma unroll
    for (int nt = 0; nt < NT; ++nt) {
      g_ushort* wrow = Wg + (unsigned)(nTile0 + nt * 64 + l15) * (unsigned)K;
      Frag b;
      b.q[0] = *(g_v4u*)(wrow + kt +      8 * g);
      b.q[1] = *(g_v4u*)(wrow + kt + 16 + 8 * g);
      acc[nt] = __builtin_amdgcn_wmma_f32_16x16x32_bf16(
          false, a.v, false, b.v, (short)0, acc[nt], false, false);
    }
  }
}

// --------------------------- weight prep (fp32 -> bf16) --------------------
__global__ void prep_weights(const float* __restrict__ Waug, const float* __restrict__ W1,
                             const float* __restrict__ W2,  const float* __restrict__ W3,
                             const float* __restrict__ Wp,  unsigned short* __restrict__ ws) {
  for (unsigned i = blockIdx.x * blockDim.x + threadIdx.x; i < W_TOTAL;
       i += gridDim.x * blockDim.x) {
    float v;
    if      (i < OFF_W1) v = Waug[i];
    else if (i < OFF_W2) v = W1[i - OFF_W1];
    else if (i < OFF_W3) v = W2[i - OFF_W2];
    else if (i < OFF_WP) v = W3[i - OFF_W3];
    else                 v = Wp[i - OFF_WP];
    ws[i] = f2bf(v);
  }
}

// ------------------------------- main kernel -------------------------------
__global__ __launch_bounds__(128, 4) void latent_ode_kernel(
    const float* __restrict__ actions, const float* __restrict__ z0,
    const float* __restrict__ s0,
    const float* __restrict__ b_aug, const float* __restrict__ b1,
    const float* __restrict__ b2,    const float* __restrict__ b3,
    const float* __restrict__ b_proj,
    const unsigned short* __restrict__ wbf, float* __restrict__ out) {

  __shared__ __attribute__((aligned(16))) unsigned short augbf[BT][SAUG];
  __shared__ __attribute__((aligned(16))) unsigned short ybf  [BT][SY];
  __shared__ __attribute__((aligned(16))) unsigned short h0   [BT][SH];
  __shared__ __attribute__((aligned(16))) unsigned short h1b  [BT][SH];

  const int tid   = threadIdx.x;
  const int lane  = tid & 31;
  const int ng    = (tid >> 5) & 3;        // wave's N group (0..3)
  const int l15   = lane & 15;
  const int g     = lane >> 4;
  const int bBase = blockIdx.x * BT;

  const unsigned short* Waug = wbf + OFF_WAUG;
  const unsigned short* W1   = wbf + OFF_W1;
  const unsigned short* W2   = wbf + OFF_W2;
  const unsigned short* W3   = wbf + OFF_W3;
  const unsigned short* Wp   = wbf + OFF_WP;

  // seed z / s regions of the aug buffer from z0, s0
  for (int i = tid; i < BT * LD; i += 128) {
    int m = i >> 7, c = i & 127;
    augbf[m][64 + c] = f2bf(z0[(bBase + m) * LD + c]);
  }
  for (int i = tid; i < BT * FS; i += 128) {
    int m = i >> 6, c = i & 63;
    augbf[m][192 + c] = f2bf(s0[(bBase + m) * FS + c]);
  }

  float y0[2][8];   // f32 master copy of RK base point (this thread's elems)
  float ks[2][8];   // RK4 weighted k accumulator

#pragma unroll 1
  for (int t = 0; t < T_DIM; ++t) {
    // ---- stage a_t into aug buffer (cols 0..63) ----
    for (int i = tid; i < BT * FA; i += 128) {
      int m = i >> 6, f = i & 63;
      augbf[m][f] = f2bf(actions[((bBase + m) * FA + f) * T_DIM + t]);
    }
    __syncthreads();   // a_t + (z,s from previous step) ready

    // ---- aug GEMM: [a|z|s](16x256) @ Waug^T -> y0 (16x128) ----
    {
      v8f acc[2];
      wave_gemm<256, 2>(&augbf[0][0], SAUG, Waug, ng * 16, acc);
#pragma unroll
      for (int ti = 0; ti < 2; ++ti) {
        int nB = ng * 16 + ti * 64;
        float bb = b_aug[nB + l15];
#pragma unroll
        for (int r = 0; r < 8; ++r) {
          float v = acc[ti][r] + bb;
          y0[ti][r] = v;
          ks[ti][r] = 0.f;
          ybf[r + 8 * g][nB + l15] = f2bf(v);
        }
      }
    }

    // ---- RK4 stages ----
#pragma unroll 1
    for (int st = 0; st < 4; ++st) {
      __syncthreads();   // ybf (stage input) ready

      // L1: y(16x128) @ W1^T -> tanh -> h0 (16x512)
      {
        v8f acc[8];
        wave_gemm<128, 8>(&ybf[0][0], SY, W1, ng * 16, acc);
#pragma unroll
        for (int ti = 0; ti < 8; ++ti) {
          int nB = ng * 16 + ti * 64;
          float bb = b1[nB + l15];
#pragma unroll
          for (int r = 0; r < 8; ++r)
            h0[r + 8 * g][nB + l15] = f2bf(fast_tanh(acc[ti][r] + bb));
        }
      }
      __syncthreads();

      // L2: h0(16x512) @ W2^T -> tanh -> h1b (16x512)
      {
        v8f acc[8];
        wave_gemm<512, 8>(&h0[0][0], SH, W2, ng * 16, acc);
#pragma unroll
        for (int ti = 0; ti < 8; ++ti) {
          int nB = ng * 16 + ti * 64;
          float bb = b2[nB + l15];
#pragma unroll
          for (int r = 0; r < 8; ++r)
            h1b[r + 8 * g][nB + l15] = f2bf(fast_tanh(acc[ti][r] + bb));
        }
      }
      __syncthreads();

      // L3: h1b(16x512) @ W3^T -> k (16x128); fused RK4 update
      {
        v8f acc[2];
        wave_gemm<512, 2>(&h1b[0][0], SH, W3, ng * 16, acc);
        const float wgt = (st == 1 || st == 2) ? 2.f : 1.f;
        const float cn  = (st < 2) ? (0.5f * DT_V) : DT_V;
#pragma unroll
        for (int ti = 0; ti < 2; ++ti) {
          int nB = ng * 16 + ti * 64;
          float bb = b3[nB + l15];
#pragma unroll
          for (int r = 0; r < 8; ++r) {
            float k = acc[ti][r] + bb;
            ks[ti][r] += wgt * k;
            int m = r + 8 * g, n = nB + l15;
            if (st < 3) {
              ybf[m][n] = f2bf(y0[ti][r] + cn * k);          // next stage input
            } else {
              float z = y0[ti][r] + (DT_V / 6.f) * ks[ti][r]; // z_{t+1}
              augbf[m][64 + n] = f2bf(z);                     // feeds proj + next aug
            }
          }
        }
      }
    }
    __syncthreads();   // z region of augbf ready

    // ---- proj: z(16x128) @ Wproj^T -> s (16x64); store out + aug s-region ----
    {
      v8f acc[1];
      wave_gemm<128, 1>(&augbf[0][64], SAUG, Wp, ng * 16, acc);
      int nB = ng * 16, n = nB + l15;
      float bb = b_proj[n];
#pragma unroll
      for (int r = 0; r < 8; ++r) {
        float s = acc[0][r] + bb;
        int m = r + 8 * g;
        out[(size_t)((bBase + m) * FS + n) * T_DIM + t] = s;
        augbf[m][192 + n] = f2bf(s);
      }
    }
    __syncthreads();
  }
}

// ------------------------------- launcher ----------------------------------
extern "C" void kernel_launch(void* const* d_in, const int* in_sizes, int n_in,
                              void* d_out, int out_size, void* d_ws, size_t ws_size,
                              hipStream_t stream) {
  const float* actions = (const float*)d_in[0];
  const float* z0      = (const float*)d_in[1];
  const float* s0      = (const float*)d_in[2];
  const float* Waug    = (const float*)d_in[3];
  const float* baug    = (const float*)d_in[4];
  const float* W1      = (const float*)d_in[5];
  const float* b1      = (const float*)d_in[6];
  const float* W2      = (const float*)d_in[7];
  const float* b2      = (const float*)d_in[8];
  const float* W3      = (const float*)d_in[9];
  const float* b3      = (const float*)d_in[10];
  const float* Wp      = (const float*)d_in[11];
  const float* bp      = (const float*)d_in[12];

  unsigned short* wbf = (unsigned short*)d_ws;   // 868,352 bytes of bf16 weights
  float* out = (float*)d_out;

  prep_weights<<<848, 512, 0, stream>>>(Waug, W1, W2, W3, Wp, wbf);
  latent_ode_kernel<<<B_DIM / BT, 128, 0, stream>>>(
      actions, z0, s0, baug, b1, b2, b3, bp, wbf, out);
}
